// GQAttention_43327630082104
// MI455X (gfx1250) — compile-verified, matrix operand order checked by reading, not run on hardware
//
#include <hip/hip_runtime.h>
#include <hip/hip_bf16.h>

// ---------------------------------------------------------------------------
// GQA attention block for gfx1250 (CDNA5, wave32, WMMA).
//   B=2, T=2048, HID=2048, NH=32, KH=8, HD=64, G=4
// Pipeline:
//   1) f32 -> f16 conversion of hidden + weights
//   2) WMMA GEMM (128x128 tile, wave tile 32x64): q/k/v projections (f16 out)
//   3) RoPE + RMSNorm (q,k) with transpose to [B,H,T,64]; V transpose
//   4) flash-attention, 64-wide KV steps (16 WMMA per step per wave)
//   5) WMMA GEMM: output projection (f32 out -> d_out)
// Non-transposed LDS staging uses GLOBAL_LOAD_ASYNC_TO_LDS_B128 (ASYNCcnt)
// when the builtin is available; next-tile global_prefetch_b8 hints added.
// ---------------------------------------------------------------------------

typedef _Float16 h16;
typedef __attribute__((ext_vector_type(16))) _Float16 v16h;
typedef __attribute__((ext_vector_type(8)))  _Float16 v8h;
typedef __attribute__((ext_vector_type(8)))  float    v8f;

#define BB   2
#define TT   2048
#define HID  2048
#define NH   32
#define KH   8
#define HD   64

#if defined(__AMDGCN__) &&                                                  \
    __has_builtin(__builtin_amdgcn_global_load_async_to_lds_b128) &&        \
    __has_builtin(__builtin_amdgcn_s_wait_asynccnt)
#define USE_ASYNC_LDS 1
#else
#define USE_ASYNC_LDS 0
#endif

// pointer types the async builtin expects: int4* in AS1 (global) / AS3 (LDS)
typedef int async_v4i __attribute__((vector_size(16)));
typedef __attribute__((address_space(1))) async_v4i* async_gptr;
typedef __attribute__((address_space(3))) async_v4i* async_lptr;

// global -> LDS 16-byte copy (async path when available)
__device__ __forceinline__ void lds_copy_b128(h16* dst, const h16* src) {
#if USE_ASYNC_LDS
  __builtin_amdgcn_global_load_async_to_lds_b128((async_gptr)src,
                                                 (async_lptr)dst, 0, 0);
#else
  *(v8h*)dst = *(const v8h*)src;
#endif
}
__device__ __forceinline__ void lds_copy_fence() {
#if USE_ASYNC_LDS
  __builtin_amdgcn_s_wait_asynccnt(0);
#endif
}

__device__ __forceinline__ v8f wmma_f16(v16h a, v16h b, v8f c) {
  // v_wmma_f32_16x16x32_f16: (neg_a, A, neg_b, B, c_mod, C, reuse_a, reuse_b)
  return __builtin_amdgcn_wmma_f32_16x16x32_f16(false, a, false, b, (short)0, c,
                                                false, false);
}

// Load one 16x32 (f16) WMMA operand fragment from an LDS tile.
// A tiles are row-major [M][stride]; B tiles are stored N-major (transposed,
// [N][stride]) so the same per-lane layout applies: lane&15 selects row/col,
// lane>>4 selects the K half (K = kh..kh+7 and 16+kh..16+kh+7).
__device__ __forceinline__ v16h load_frag(const h16* base, int row0, int stride,
                                          int koff) {
  const int lane = threadIdx.x & 31;
  const int r    = lane & 15;
  const int kh   = (lane >> 4) << 3;
  const h16* p = base + (size_t)(row0 + r) * stride + koff + kh;
  union { v16h v; v8h h[2]; } u;
  u.h[0] = *(const v8h*)(p);
  u.h[1] = *(const v8h*)(p + 16);
  return u.v;
}

// ---------------------------------------------------------------------------
// 1) f32 -> f16 conversion
// ---------------------------------------------------------------------------
__global__ __launch_bounds__(256) void cvt_f32_to_f16(const float* __restrict__ in,
                                                      h16* __restrict__ out, int n) {
  int i = (blockIdx.x * 256 + threadIdx.x) * 4;
  if (i < n) {
#pragma unroll
    for (int j = 0; j < 4; ++j) out[i + j] = (h16)in[i + j];
  }
}

// ---------------------------------------------------------------------------
// 2/5) WMMA GEMM: C[M,N] = A[M,K] * Bm[K,N], A/B f16, C f16 or f32.
//      Workgroup tile 128x128, 8 waves (4x2), wave tile 32x64 (2x4 frags).
// ---------------------------------------------------------------------------
template <bool OUT_F16>
__global__ __launch_bounds__(256) void gemm_wmma(const h16* __restrict__ A,
                                                 const h16* __restrict__ Bm,
                                                 void* __restrict__ Cout,
                                                 int M, int N, int K) {
  __shared__ __align__(16) h16 sA[128 * 32];   // [m][k]
  __shared__ __align__(16) h16 sB[128 * 32];   // [n][k] (transposed)

  const int tid  = threadIdx.x;
  const int wave = tid >> 5;
  const int lane = tid & 31;
  const int hl   = lane >> 4;
  const int ln   = lane & 15;

  const int m0 = blockIdx.y * 128;
  const int n0 = blockIdx.x * 128;
  const int wm = (wave >> 1) * 32;   // wave row within tile
  const int wn = (wave & 1) * 64;    // wave col within tile

  v8f acc[2][4] = {};

  for (int k0 = 0; k0 < K; k0 += 32) {
    // stage A tile 128x32 (row-major) -- async global->LDS
#pragma unroll
    for (int i = 0; i < 2; ++i) {
      int c = tid + 256 * i;               // 512 chunks of 8 halves
      int row = c >> 2, off = (c & 3) * 8;
      lds_copy_b128(&sA[row * 32 + off],
                    &A[(size_t)(m0 + row) * K + k0 + off]);
      if (k0 + 32 < K)
        __builtin_prefetch(&A[(size_t)(m0 + row) * K + k0 + 32 + off], 0, 1);
    }
    // stage B tile 32x128 transposed -> sB[n][k] (needs VGPR round-trip)
#pragma unroll
    for (int i = 0; i < 2; ++i) {
      int c = tid + 256 * i;               // 512 chunks of 8 halves
      int k = c >> 4, n8 = (c & 15) * 8;
      v8h b = *(const v8h*)&Bm[(size_t)(k0 + k) * N + n0 + n8];
#pragma unroll
      for (int j = 0; j < 8; ++j) sB[(n8 + j) * 32 + k] = b[j];
      if (k0 + 32 < K)
        __builtin_prefetch(&Bm[(size_t)(k0 + 32 + k) * N + n0 + n8], 0, 1);
    }
    lds_copy_fence();
    __syncthreads();

    v16h af[2], bf[4];
    af[0] = load_frag(sA, wm + 0, 32, 0);
    af[1] = load_frag(sA, wm + 16, 32, 0);
#pragma unroll
    for (int fn = 0; fn < 4; ++fn) bf[fn] = load_frag(sB, wn + fn * 16, 32, 0);
#pragma unroll
    for (int fm = 0; fm < 2; ++fm)
#pragma unroll
      for (int fn = 0; fn < 4; ++fn)
        acc[fm][fn] = wmma_f16(af[fm], bf[fn], acc[fm][fn]);
    __syncthreads();
  }

  // epilogue: C fragment layout: vgpr r -> row r + 8*hl, lane&15 -> col
#pragma unroll
  for (int fm = 0; fm < 2; ++fm)
#pragma unroll
    for (int fn = 0; fn < 4; ++fn)
#pragma unroll
      for (int r = 0; r < 8; ++r) {
        int row = m0 + wm + fm * 16 + r + 8 * hl;
        int col = n0 + wn + fn * 16 + ln;
        float v = acc[fm][fn][r];
        if (OUT_F16)
          ((h16*)Cout)[(size_t)row * N + col] = (h16)v;
        else
          ((float*)Cout)[(size_t)row * N + col] = v;
      }
}

// ---------------------------------------------------------------------------
// 3) RoPE + RMSNorm.  One wave per (b,t,h) row of 64 values; lane j handles
//    the interleaved pair (2j, 2j+1).  Input [B,T,H,64] -> output [B,H,T,64].
// ---------------------------------------------------------------------------
__global__ __launch_bounds__(256) void rope_rmsnorm(const h16* __restrict__ in,
                                                    h16* __restrict__ out,
                                                    const float* __restrict__ w,
                                                    int heads) {
  const int tid  = threadIdx.x;
  const int wave = tid >> 5;
  const int lane = tid & 31;
  const int rid  = blockIdx.x * 8 + wave;          // ((b*T + t)*heads + h)
  const int h    = rid % heads;
  const int t    = (rid / heads) % TT;
  const int b    = rid / (heads * TT);

  const h16* src = in + (size_t)rid * HD;
  float x1 = (float)src[2 * lane];
  float x2 = (float)src[2 * lane + 1];

  float inv_freq = __powf(10000.0f, -(float)lane / 32.0f);
  float f = (float)t * inv_freq;
  float sn, cs;
  __sincosf(f, &sn, &cs);
  float r1 = x1 * cs - x2 * sn;
  float r2 = x1 * sn + x2 * cs;

  float ss = r1 * r1 + r2 * r2;
#pragma unroll
  for (int off = 1; off < 32; off <<= 1) ss += __shfl_xor(ss, off, 32);
  float sc = rsqrtf(ss * (1.0f / 64.0f) + 1e-6f);

  r1 *= sc * w[2 * lane];
  r2 *= sc * w[2 * lane + 1];

  h16* dst = out + ((size_t)(b * heads + h) * TT + t) * HD;
  dst[2 * lane]     = (h16)r1;
  dst[2 * lane + 1] = (h16)r2;
}

// V: [B,T,KH,64] -> [B,KH,T,64], 8 halves per thread
__global__ __launch_bounds__(256) void transpose_v(const h16* __restrict__ in,
                                                   h16* __restrict__ out) {
  int c  = blockIdx.x * 256 + threadIdx.x;   // B*T*KH*8 chunks
  int d8 = (c & 7) * 8;
  int h  = (c >> 3) & (KH - 1);
  int t  = (c >> 6) & (TT - 1);
  int b  = c >> 17;
  v8h v = *(const v8h*)&in[(((size_t)(b * TT + t) * KH) + h) * HD + d8];
  *(v8h*)&out[(((size_t)(b * KH + h) * TT) + t) * HD + d8] = v;
}

// ---------------------------------------------------------------------------
// 4) Flash attention.  grid = (T/128, B*NH).  8 waves, 16 Q rows per wave,
//    KV processed in 64-wide steps (16 WMMA per step per wave).
//    q16/k16/v16 in [B,H,T,64] f16; ctx out in [B,T,NH*64] f16.
// ---------------------------------------------------------------------------
__global__ __launch_bounds__(256) void attn_flash(const h16* __restrict__ q16,
                                                  const h16* __restrict__ k16,
                                                  const h16* __restrict__ v16,
                                                  h16* __restrict__ ctx) {
  __shared__ __align__(16) h16 sQ[128 * 64];
  __shared__ __align__(16) h16 sK[64 * 64];    // [kv][d] == B^T for Q*K^T
  __shared__ __align__(16) h16 sVt[64 * 64];   // [d][kv] == B^T for P*V
  __shared__ __align__(16) h16 sP[8][16 * 64]; // per-wave P staging

  const int tid  = threadIdx.x;
  const int wave = tid >> 5;
  const int lane = tid & 31;
  const int hl   = lane >> 4;
  const int ln   = lane & 15;

  const int q0  = blockIdx.x * 128;
  const int bh  = blockIdx.y;            // b*NH + qh
  const int b   = bh >> 5;
  const int qh  = bh & 31;
  const int kvh = qh >> 2;               // G = 4

  const size_t qbase = ((size_t)(b * NH + qh) * TT + q0) * HD;
  const size_t kbase = ((size_t)(b * KH + kvh) * TT) * HD;

  // stage the 128x64 Q tile (async)
#pragma unroll
  for (int i = 0; i < 4; ++i) {
    int c = tid + 256 * i;
    int row = c >> 3, off = (c & 7) * 8;
    lds_copy_b128(&sQ[row * 64 + off], &q16[qbase + (size_t)row * 64 + off]);
  }
  lds_copy_fence();
  __syncthreads();

  // resident Q fragments for this wave's 16 rows (two K-chunks of 32)
  v16h aq[2];
  aq[0] = load_frag(sQ, wave * 16, 64, 0);
  aq[1] = load_frag(sQ, wave * 16, 64, 32);

  v8f acc[4] = {};
  float run_max[8], run_sum[8];
#pragma unroll
  for (int r = 0; r < 8; ++r) { run_max[r] = -__builtin_inff(); run_sum[r] = 0.f; }

  const int q_hi = q0 + 127;
  for (int kv0 = 0; kv0 <= q_hi; kv0 += 64) {
    // stage K tile 64x64 (natural, async) and V tile (transposed)
#pragma unroll
    for (int i = 0; i < 2; ++i) {
      int c = tid + 256 * i;               // 512 chunks of 8 halves
      int row = c >> 3, off = (c & 7) * 8;
      lds_copy_b128(&sK[row * 64 + off],
                    &k16[kbase + (size_t)(kv0 + row) * 64 + off]);
      v8h vv = *(const v8h*)&v16[kbase + (size_t)(kv0 + row) * 64 + off];
#pragma unroll
      for (int j = 0; j < 8; ++j) sVt[(off + j) * 64 + row] = vv[j];
    }
    lds_copy_fence();
    __syncthreads();

    // S = Q * K^T : 16x64, four 16x16 fragments, K-dim = 64
    v8f s[4] = {};
#pragma unroll
    for (int jn = 0; jn < 4; ++jn) {
      v16h bk0 = load_frag(sK, jn * 16, 64, 0);
      v16h bk1 = load_frag(sK, jn * 16, 64, 32);
      s[jn] = wmma_f16(aq[0], bk0, s[jn]);
      s[jn] = wmma_f16(aq[1], bk1, s[jn]);
    }

    // online softmax in C-fragment domain
    const float scale = 0.125f;   // 1/sqrt(64)
    float p[4][8], corr[8];
#pragma unroll
    for (int r = 0; r < 8; ++r) {
      int qg = q0 + wave * 16 + r + 8 * hl;
      float mx = -__builtin_inff();
#pragma unroll
      for (int jn = 0; jn < 4; ++jn) {
        int kg = kv0 + jn * 16 + ln;
        float v = s[jn][r] * scale;
        v = (kg <= qg) ? v : -__builtin_inff();
        p[jn][r] = v;
        mx = fmaxf(mx, v);
      }
#pragma unroll
      for (int off = 1; off < 16; off <<= 1)
        mx = fmaxf(mx, __shfl_xor(mx, off, 32));
      float nm = fmaxf(run_max[r], mx);
      corr[r] = __expf(run_max[r] - nm);
      run_max[r] = nm;
      float rs = 0.f;
#pragma unroll
      for (int jn = 0; jn < 4; ++jn) {
        float e = __expf(p[jn][r] - nm);
        p[jn][r] = e;
        rs += e;
      }
#pragma unroll
      for (int off = 1; off < 16; off <<= 1) rs += __shfl_xor(rs, off, 32);
      run_sum[r] = run_sum[r] * corr[r] + rs;
    }

    // rescale running accumulators
#pragma unroll
    for (int d = 0; d < 4; ++d)
#pragma unroll
      for (int r = 0; r < 8; ++r) acc[d][r] *= corr[r];

    // C-layout -> A-layout via per-wave LDS staging (16x64 P tile)
    h16* myP = &sP[wave][0];
#pragma unroll
    for (int jn = 0; jn < 4; ++jn)
#pragma unroll
      for (int r = 0; r < 8; ++r)
        myP[(r + 8 * hl) * 64 + jn * 16 + ln] = (h16)p[jn][r];
    v16h ap[2];
    ap[0] = load_frag(myP, 0, 64, 0);
    ap[1] = load_frag(myP, 0, 64, 32);

    // O += P * V  (four 16-wide d chunks, two 32-deep kv chunks)
#pragma unroll
    for (int d = 0; d < 4; ++d) {
      v16h bv0 = load_frag(sVt, d * 16, 64, 0);
      v16h bv1 = load_frag(sVt, d * 16, 64, 32);
      acc[d] = wmma_f16(ap[0], bv0, acc[d]);
      acc[d] = wmma_f16(ap[1], bv1, acc[d]);
    }
    __syncthreads();
  }

  // epilogue: normalize, store ctx in [B,T,NH*64]
#pragma unroll
  for (int r = 0; r < 8; ++r) {
    int qg = q0 + wave * 16 + r + 8 * hl;
    float inv = 1.0f / run_sum[r];
#pragma unroll
    for (int d = 0; d < 4; ++d) {
      ctx[((size_t)(b * TT + qg) * (NH * HD)) + qh * HD + d * 16 + ln] =
          (h16)(acc[d][r] * inv);
    }
  }
}

// ---------------------------------------------------------------------------
// host launcher
// ---------------------------------------------------------------------------
extern "C" void kernel_launch(void* const* d_in, const int* in_sizes, int n_in,
                              void* d_out, int out_size, void* d_ws, size_t ws_size,
                              hipStream_t stream) {
  const float* hs = (const float*)d_in[0];
  const float* wq = (const float*)d_in[1];
  const float* wk = (const float*)d_in[2];
  const float* wv = (const float*)d_in[3];
  const float* wo = (const float*)d_in[4];
  const float* qn = (const float*)d_in[5];
  const float* kn = (const float*)d_in[6];

  const size_t M   = (size_t)BB * TT;       // 4096
  const size_t KVN = (size_t)KH * HD;       // 512

  h16* ws = (h16*)d_ws;
  size_t o = 0;
  h16* hs16 = ws + o; o += M * HID;
  h16* wq16 = ws + o; o += (size_t)HID * HID;
  h16* wk16 = ws + o; o += (size_t)HID * KVN;
  h16* wv16 = ws + o; o += (size_t)HID * KVN;
  h16* wo16 = ws + o; o += (size_t)HID * HID;
  h16* qp   = ws + o; o += M * HID;         // [B,T,NH,64]
  h16* kp   = ws + o; o += M * KVN;         // [B,T,KH,64]
  h16* vp   = ws + o; o += M * KVN;
  h16* q16  = ws + o; o += M * HID;         // [B,NH,T,64]
  h16* k16  = ws + o; o += M * KVN;         // [B,KH,T,64]
  h16* v16  = ws + o; o += M * KVN;
  h16* ctx  = ws + o; o += M * HID;         // [B,T,NH*64]

  // 1) conversions
  auto cvt = [&](const float* src, h16* dst, size_t n) {
    cvt_f32_to_f16<<<dim3((unsigned)(n / 1024)), dim3(256), 0, stream>>>(src, dst, (int)n);
  };
  cvt(hs, hs16, M * HID);
  cvt(wq, wq16, (size_t)HID * HID);
  cvt(wk, wk16, (size_t)HID * KVN);
  cvt(wv, wv16, (size_t)HID * KVN);
  cvt(wo, wo16, (size_t)HID * HID);

  // 2) QKV projections (f16 out)
  gemm_wmma<true><<<dim3(HID / 128, M / 128), dim3(256), 0, stream>>>(
      hs16, wq16, qp, (int)M, HID, HID);
  gemm_wmma<true><<<dim3((unsigned)(KVN / 128), M / 128), dim3(256), 0, stream>>>(
      hs16, wk16, kp, (int)M, (int)KVN, HID);
  gemm_wmma<true><<<dim3((unsigned)(KVN / 128), M / 128), dim3(256), 0, stream>>>(
      hs16, wv16, vp, (int)M, (int)KVN, HID);

  // 3) RoPE + RMSNorm (+ transpose to [B,H,T,64]); V transpose
  rope_rmsnorm<<<dim3((unsigned)(BB * TT * NH / 8)), dim3(256), 0, stream>>>(qp, q16, qn, NH);
  rope_rmsnorm<<<dim3((unsigned)(BB * TT * KH / 8)), dim3(256), 0, stream>>>(kp, k16, kn, KH);
  transpose_v<<<dim3((unsigned)(BB * TT * KH * 8 / 256)), dim3(256), 0, stream>>>(vp, v16);

  // 4) attention
  attn_flash<<<dim3(TT / 128, BB * NH), dim3(256), 0, stream>>>(q16, k16, v16, ctx);

  // 5) output projection (f32 out -> d_out)
  gemm_wmma<false><<<dim3(HID / 128, M / 128), dim3(256), 0, stream>>>(
      ctx, wo16, d_out, (int)M, HID, HID);
}
